// lstm_tagger_4767413699078
// MI455X (gfx1250) — matrix operand order repaired and checked by reading, not compiled, not run on hardware
//
#include <hip/hip_runtime.h>

// ---------------------------------------------------------------------------
// LSTM tagger for MI455X (gfx1250, wave32, WMMA + TDM).
//  S=8192 words, L=16 chars, CHAR_DIM=64, WORD_DIM=256, CHAR_HID=128,
//  WORD_HID=512, N_TAG=64.
// ---------------------------------------------------------------------------

typedef __attribute__((ext_vector_type(16))) __bf16 v16bf;
typedef __attribute__((ext_vector_type(8)))  float  v8f;
typedef __attribute__((ext_vector_type(4)))  unsigned u32x4;
typedef __attribute__((ext_vector_type(8)))  int      i32x8;
typedef __attribute__((ext_vector_type(4)))  int      i32x4;

union Frag { unsigned u[8]; v16bf v; };

// A fragment (16x32 bf16, M x K), per ISA 7.12.2:
// lanes 0-15: row=lane, v0..3 K=0..7, v4..7 K=16..23
// lanes 16-31: row=lane-16, v0..3 K=8..15, v4..7 K=24..31
__device__ __forceinline__ v16bf load_a_frag(const __bf16* A, int lda, int row0,
                                             int k0, int lane) {
  int m  = lane & 15;
  int kh = (lane < 16) ? 0 : 4;  // uint offset inside K-half
  const unsigned* base = (const unsigned*)(A + (size_t)(row0 + m) * lda + k0);
  Frag f;
#pragma unroll
  for (int i = 0; i < 4; ++i) { f.u[i] = base[kh + i]; f.u[4 + i] = base[8 + kh + i]; }
  return f.v;
}

// B fragment (32x16 bf16, K x N) from row-major W[N][K] (i.e. B = W^T):
// lanes 0-15: n=lane, K=0..7,16..23 ; lanes 16-31: n=lane-16, K=8..15,24..31
__device__ __forceinline__ v16bf load_b_frag(const __bf16* W, int ldw, int n0,
                                             int k0, int lane) {
  int n  = lane & 15;
  int kb = (lane < 16) ? 0 : 8;
  const unsigned* base = (const unsigned*)(W + (size_t)(n0 + n) * ldw + k0 + kb);
  Frag f;
#pragma unroll
  for (int i = 0; i < 4; ++i) { f.u[i] = base[i]; f.u[4 + i] = base[8 + i]; }
  return f.v;
}

__device__ __forceinline__ float sigm(float x)   { return 1.f / (1.f + __expf(-x)); }
__device__ __forceinline__ float tanh_f(float x) { return 2.f / (1.f + __expf(-2.f * x)) - 1.f; }

// ---------------------------------------------------------------------------
// TDM: 2D tile load (bf16, rows x 32 cols) global -> LDS, per ISA D# layout.
//  group0: [1:0]=count=1, [63:32]=lds byte addr, [120:64]=global addr,
//          [127:126]=type=2
//  group1: [17:16]=data_size(1 => 2B), tensor_dim0 [79:48], tensor_dim1
//          [111:80], tile_dim0 [127:112], tile_dim1 [143:128],
//          tensor_dim0_stride [207:160]
// Issued by one wave; completion tracked with TENSORcnt.
// ---------------------------------------------------------------------------
__device__ __forceinline__ void tdm_load_tile_bf16(const __bf16* gsrc,
                                                   void* ldst, int ld_elems,
                                                   int rows, int tensor_rows,
                                                   int tensor_cols) {
  unsigned long long ga = (unsigned long long)(size_t)gsrc;
  unsigned lo = (unsigned)((size_t)ldst & 0xFFFFFFFFull);  // LDS byte offset
  u32x4 g0;
  g0.x = 1u;                                       // count=1, user descriptor
  g0.y = lo;                                       // lds_addr
  g0.z = (unsigned)(ga & 0xFFFFFFFFull);           // global_addr[31:0]
  g0.w = (unsigned)((ga >> 32) & 0x01FFFFFFull) | (2u << 30);  // [56:32], type=2
  i32x8 g1;
  g1[0] = 1 << 16;                                 // data_size = 2 bytes
  g1[1] = (tensor_cols & 0xFFFF) << 16;            // tensor_dim0[15:0]
  g1[2] = ((tensor_cols >> 16) & 0xFFFF) | ((tensor_rows & 0xFFFF) << 16);
  g1[3] = ((tensor_rows >> 16) & 0xFFFF) | (32 << 16);   // tile_dim0 = 32
  g1[4] = rows & 0xFFFF;                           // tile_dim1, tile_dim2=0
  g1[5] = ld_elems;                                // tensor_dim0_stride[31:0]
  g1[6] = 0;
  g1[7] = 0;
  i32x4 z4 = {0, 0, 0, 0};
  i32x8 z8 = {0, 0, 0, 0, 0, 0, 0, 0};
  __builtin_amdgcn_tensor_load_to_lds(g0, g1, z4, z4, z8, 0);
}

// ---------------------------------------------------------------------------
// Workspace layout (bytes)
// ---------------------------------------------------------------------------
static constexpr size_t OFF_CW    = 0;                         // 512*192 bf16 char [Wih|Whh]
static constexpr size_t OFF_CBIAS = OFF_CW    + 512*192*2;     // 512 f32
static constexpr size_t OFF_WWIH  = OFF_CBIAS + 512*4;         // 2048*384 bf16
static constexpr size_t OFF_WBIAS = OFF_WWIH  + 2048*384*2;    // 2048 f32
static constexpr size_t OFF_CLSW  = OFF_WBIAS + 2048*4;        // 64*512 bf16
static constexpr size_t OFF_FEATS = OFF_CLSW  + 64*512*2;      // 8192*384 bf16
static constexpr size_t OFF_WPRE  = OFF_FEATS + (size_t)8192*384*2;  // 8192*2048 f32
static constexpr size_t OFF_HS    = OFF_WPRE  + (size_t)8192*2048*4; // 8192*512 f32
static constexpr size_t OFF_HSB   = OFF_HS    + (size_t)8192*512*4;  // 8192*512 bf16
static constexpr size_t OFF_CTR   = OFF_HSB   + (size_t)8192*512*2;  // sync counter

// ---------------------------------------------------------------------------
// Prep kernels: fp32 -> bf16 weight conversion, bias sums, word-embed gather
// ---------------------------------------------------------------------------
__global__ void k_prep_charw(const float* cWih, const float* cWhh,
                             const float* cbih, const float* cbhh,
                             __bf16* cw, float* cb) {
  int i = blockIdx.x * blockDim.x + threadIdx.x;
  if (i < 512 * 192) {
    int g = i / 192, k = i % 192;
    float v = (k < 64) ? cWih[g * 64 + k] : cWhh[g * 128 + (k - 64)];
    cw[i] = (__bf16)v;
  }
  if (i < 512) cb[i] = cbih[i] + cbhh[i];
}

__global__ void k_prep_wordw(const float* wWih, const float* wbih,
                             const float* wbhh, __bf16* ww, float* wb) {
  int i = blockIdx.x * blockDim.x + threadIdx.x;
  if (i < 2048 * 384) ww[i] = (__bf16)wWih[i];
  if (i < 2048) wb[i] = wbih[i] + wbhh[i];
}

__global__ void k_prep_cls(const float* clsW, __bf16* cb) {
  int i = blockIdx.x * blockDim.x + threadIdx.x;
  if (i < 64 * 512) cb[i] = (__bf16)clsW[i];
}

__global__ void k_embed_words(const int* x, const float* wemb, __bf16* feats) {
  int i = blockIdx.x * blockDim.x + threadIdx.x;  // 8192*256
  int s = i >> 8, d = i & 255;
  feats[(size_t)s * 384 + d] = (__bf16)wemb[(size_t)x[s] * 256 + d];
}

// ---------------------------------------------------------------------------
// Char LSTM: one block = 16 words, entire 16-step recurrence in LDS.
// gates(16x512) = [x_t | h](16x192) @ Wc^T(192x512) + b, via WMMA bf16.
// Wave w owns hidden cols [16w,16w+16): gate tiles {w, 8+w, 16+w, 24+w}.
// ---------------------------------------------------------------------------
__global__ __launch_bounds__(256) void k_char_lstm(
    const int* chars, const int* lens, const float* char_embed,
    const __bf16* cw, const float* cbias, __bf16* feats) {
  extern __shared__ unsigned char smem[];
  __bf16* Ws   = (__bf16*)smem;               // 512 x 192 bf16   (196608 B)
  __bf16* Ast  = (__bf16*)(smem + 196608);    // 16 x 192 bf16    (6144 B)
  __bf16* cet  = (__bf16*)(smem + 202752);    // 128 x 64 bf16    (16384 B)
  float*  bS   = (float*)(smem + 219136);     // 512 f32          (2048 B)
  int*    lenS = (int*)(smem + 221184);       // 16 int           (64 B)

  const int tid = threadIdx.x, lane = tid & 31, w = tid >> 5;
  const int word0 = blockIdx.x * 16;

  {  // stage weights (bf16, uint4 copies), char-embed table, bias, lens
    const uint4* src = (const uint4*)cw;
    uint4* dst = (uint4*)Ws;
    for (int i = tid; i < (512 * 192 * 2) / 16; i += 256) dst[i] = src[i];
    for (int i = tid; i < 128 * 64; i += 256) cet[i] = (__bf16)char_embed[i];
    for (int i = tid; i < 512; i += 256) bS[i] = cbias[i];
    if (tid < 16) lenS[tid] = lens[word0 + tid];
    for (int i = tid; i < 16 * 128; i += 256) {  // h := 0 in staging cols 64..191
      int r = i >> 7, c = i & 127;
      Ast[r * 192 + 64 + c] = (__bf16)0.f;
    }
  }
  __syncthreads();

  const int g0 = 16 * w;            // hidden-unit base for this wave
  const int nc = lane & 15;
  float bI = bS[g0 + nc], bF = bS[128 + g0 + nc];
  float bG = bS[256 + g0 + nc], bO = bS[384 + g0 + nc];

  v8f cst;                          // c state, one 16x16 fragment per wave
#pragma unroll
  for (int v = 0; v < 8; ++v) cst[v] = 0.f;

#pragma unroll 1
  for (int t = 0; t < 16; ++t) {
    {  // stage x_t: 16 words x 64 cols bf16 from embed table (8B copies)
      int wrd = tid >> 4, cg = (tid & 15) * 4;
      int ch = chars[(word0 + wrd) * 16 + t];
      *(unsigned long long*)(Ast + wrd * 192 + cg) =
          *(const unsigned long long*)(cet + ch * 64 + cg);
    }
    __syncthreads();

    v8f aI, aF, aG, aO;
#pragma unroll
    for (int v = 0; v < 8; ++v) { aI[v] = bI; aF[v] = bF; aG[v] = bG; aO[v] = bO; }

#pragma unroll
    for (int kc = 0; kc < 6; ++kc) {           // K = 192 = 6 x 32
      v16bf a = load_a_frag(Ast, 192, 0, kc * 32, lane);
      v16bf b0 = load_b_frag(Ws, 192, g0,       kc * 32, lane);
      v16bf b1 = load_b_frag(Ws, 192, 128 + g0, kc * 32, lane);
      v16bf b2 = load_b_frag(Ws, 192, 256 + g0, kc * 32, lane);
      v16bf b3 = load_b_frag(Ws, 192, 384 + g0, kc * 32, lane);
      aI = __builtin_amdgcn_wmma_f32_16x16x32_bf16(false, a, false, b0, (short)0, aI, false, false);
      aF = __builtin_amdgcn_wmma_f32_16x16x32_bf16(false, a, false, b1, (short)0, aF, false, false);
      aG = __builtin_amdgcn_wmma_f32_16x16x32_bf16(false, a, false, b2, (short)0, aG, false, false);
      aO = __builtin_amdgcn_wmma_f32_16x16x32_bf16(false, a, false, b3, (short)0, aO, false, false);
    }

    float hv[8];
#pragma unroll
    for (int v = 0; v < 8; ++v) {
      float iv = sigm(aI[v]), fv = sigm(aF[v]);
      float gv = tanh_f(aG[v]), ov = sigm(aO[v]);
      float cv = fv * cst[v] + iv * gv;
      cst[v] = cv;
      hv[v] = ov * tanh_f(cv);
      int row = v + 8 * (lane >> 4);
      if (lenS[row] - 1 == t)  // last valid char of this word -> char_feat
        feats[(size_t)(word0 + row) * 384 + 256 + g0 + nc] = (__bf16)hv[v];
    }
    __syncthreads();  // all A reads of step t done before h overwrite
#pragma unroll
    for (int v = 0; v < 8; ++v) {
      int row = v + 8 * (lane >> 4);
      Ast[row * 192 + 64 + g0 + nc] = (__bf16)hv[v];
    }
  }
}

// ---------------------------------------------------------------------------
// WMMA GEMM with TDM double-buffered LDS staging:
//   out[M,N] = A[M,K](bf16) @ W[N,K]^T(bf16) + bias (f32)
// Block = 128-row x 64-col tile, 8 waves x 16 rows, 4 N-tiles per wave.
// Per 32-wide K chunk, wave 0 issues TENSOR_LOAD_TO_LDS for the A tile
// (128x32) and B tile (64x32) of chunk c+1 while all waves run WMMA on
// chunk c out of LDS; sync via s_wait_tensorcnt + workgroup barrier.
// ---------------------------------------------------------------------------
__global__ __launch_bounds__(256) void k_gemm_bias(
    const __bf16* __restrict__ A, const __bf16* __restrict__ W,
    const float* __restrict__ bias, float* __restrict__ out,
    int M, int N, int K, int lda, int ldw, int ldo) {
  extern __shared__ unsigned char smem[];
  __bf16* Ast = (__bf16*)smem;                 // 2 x (128 x 32) bf16 = 16384 B
  __bf16* Bst = (__bf16*)(smem + 16384);       // 2 x (64 x 32)  bf16 =  8192 B

  const int lane = threadIdx.x & 31, w = threadIdx.x >> 5;
  const int row0 = blockIdx.x * 128;
  const int n0 = blockIdx.y * 64;
  const int nc = lane & 15;
  const int nchunk = K >> 5;
  const bool tdmWave = (threadIdx.x < 32);     // wave 0 drives the TDM

  v8f acc[4];
#pragma unroll
  for (int nt = 0; nt < 4; ++nt) {
    float b = bias[n0 + nt * 16 + nc];
#pragma unroll
    for (int v = 0; v < 8; ++v) acc[nt][v] = b;
  }

  if (tdmWave) {  // prefetch chunk 0
    tdm_load_tile_bf16(A + (size_t)row0 * lda, Ast, lda, 128, M, K);
    tdm_load_tile_bf16(W + (size_t)n0 * ldw, Bst, ldw, 64, N, K);
  }

#pragma unroll 1
  for (int c = 0; c < nchunk; ++c) {
    if (tdmWave) {
      if (c + 1 < nchunk) {  // issue chunk c+1 into the other buffer
        int k1 = (c + 1) * 32, nb = (c + 1) & 1;
        tdm_load_tile_bf16(A + (size_t)row0 * lda + k1, Ast + nb * 128 * 32,
                           lda, 128, M, K);
        tdm_load_tile_bf16(W + (size_t)n0 * ldw + k1, Bst + nb * 64 * 32,
                           ldw, 64, N, K);
        __builtin_amdgcn_s_wait_tensorcnt(2);  // chunk c (first pair) done
      } else {
        __builtin_amdgcn_s_wait_tensorcnt(0);
      }
    }
    __syncthreads();  // staged tiles visible to all waves

    const __bf16* Ab = Ast + (c & 1) * 128 * 32;
    const __bf16* Bb = Bst + (c & 1) * 64 * 32;
    v16bf a = load_a_frag(Ab, 32, w * 16, 0, lane);
#pragma unroll
    for (int nt = 0; nt < 4; ++nt) {
      v16bf b = load_b_frag(Bb, 32, nt * 16, 0, lane);
      acc[nt] = __builtin_amdgcn_wmma_f32_16x16x32_bf16(false, a, false, b,
                                                        (short)0, acc[nt], false, false);
    }
    __syncthreads();  // all reads done before buffer reuse at c+2
  }

#pragma unroll
  for (int nt = 0; nt < 4; ++nt)
#pragma unroll
    for (int v = 0; v < 8; ++v) {
      int row = row0 + w * 16 + v + 8 * (lane >> 4);
      out[(size_t)row * ldo + n0 + nt * 16 + nc] = acc[nt][v];
    }
}

// ---------------------------------------------------------------------------
// Word LSTM recurrence: 16 persistent blocks, block b owns hidden [32b,32b+32)
// (128 gate rows). W_hh slice (128x512 f32, padded stride 520) lives in LDS.
// Per step: gemv from LDS, activations, write h slice, agent-scope barrier.
// ---------------------------------------------------------------------------
__global__ __launch_bounds__(256) void k_word_lstm(
    const float* __restrict__ wWhh, const float* __restrict__ wpre,
    float* __restrict__ hs, __bf16* __restrict__ hsb, unsigned* ctr) {
  extern __shared__ unsigned char smem[];
  float* Ws = (float*)smem;                       // 128 x 520 f32
  float* hL = (float*)(smem + 128 * 520 * 4);     // 512 f32
  float* gL = hL + 512;                           // 128 f32
  float* cL = gL + 128;                           // 32 f32

  const int tid = threadIdx.x;
  const int b = blockIdx.x;

  for (int i = tid; i < 128 * 512; i += 256) {    // stage W_hh slice
    int r = i >> 9, k = i & 511;
    int G = (r >> 5) * 512 + 32 * b + (r & 31);   // global gate row
    Ws[r * 520 + k] = wWhh[(size_t)G * 512 + k];
  }
  if (tid < 32) cL[tid] = 0.f;
  __syncthreads();

  const int r = tid >> 1, half = tid & 1, kb = half * 256;
  const int G = (r >> 5) * 512 + 32 * b + (r & 31);

#pragma unroll 1
  for (int t = 0; t < 8192; ++t) {
    for (int k = tid; k < 512; k += 256)          // stage h_{t-1}
      hL[k] = (t == 0) ? 0.f : hs[(size_t)(t - 1) * 512 + k];
    __syncthreads();

    const float4* wp = (const float4*)(Ws + r * 520 + kb);
    const float4* hp = (const float4*)(hL + kb);
    float acc = 0.f;
#pragma unroll 8
    for (int i = 0; i < 64; ++i) {
      float4 a = wp[i], h4 = hp[i];
      acc = fmaf(a.x, h4.x, acc); acc = fmaf(a.y, h4.y, acc);
      acc = fmaf(a.z, h4.z, acc); acc = fmaf(a.w, h4.w, acc);
    }
    acc += __shfl_xor(acc, 1, 32);                // pair-reduce halves
    if (half == 0) gL[r] = acc + wpre[(size_t)t * 2048 + G];
    __syncthreads();

    if (tid < 32) {
      float iv = sigm(gL[tid]),      fv = sigm(gL[32 + tid]);
      float gv = tanh_f(gL[64 + tid]), ov = sigm(gL[96 + tid]);
      float cv = fv * cL[tid] + iv * gv;
      cL[tid] = cv;
      float hv = ov * tanh_f(cv);
      int col = 32 * b + tid;
      hs[(size_t)t * 512 + col] = hv;
      hsb[(size_t)t * 512 + col] = (__bf16)hv;
    }
    __builtin_amdgcn_fence(__ATOMIC_RELEASE, "agent");
    __syncthreads();
    if (t < 8191) {                               // cross-block step barrier
      if (tid == 0) {
        __hip_atomic_fetch_add(ctr, 1u, __ATOMIC_RELEASE, __HIP_MEMORY_SCOPE_AGENT);
        while (__hip_atomic_load(ctr, __ATOMIC_ACQUIRE, __HIP_MEMORY_SCOPE_AGENT) <
               16u * (unsigned)(t + 1)) { }
      }
      __syncthreads();
      __builtin_amdgcn_fence(__ATOMIC_ACQUIRE, "agent");
    }
  }
}

// ---------------------------------------------------------------------------
extern "C" void kernel_launch(void* const* d_in, const int* in_sizes, int n_in,
                              void* d_out, int out_size, void* d_ws, size_t ws_size,
                              hipStream_t stream) {
  const int*   x     = (const int*)d_in[0];
  const int*   chars = (const int*)d_in[1];
  const int*   lens  = (const int*)d_in[2];
  const float* wemb  = (const float*)d_in[3];
  const float* cemb  = (const float*)d_in[4];
  const float* cWih  = (const float*)d_in[5];
  const float* cWhh  = (const float*)d_in[6];
  const float* cbih  = (const float*)d_in[7];
  const float* cbhh  = (const float*)d_in[8];
  const float* wWih  = (const float*)d_in[9];
  const float* wWhh  = (const float*)d_in[10];
  const float* wbih  = (const float*)d_in[11];
  const float* wbhh  = (const float*)d_in[12];
  const float* clsW  = (const float*)d_in[13];
  const float* clsb  = (const float*)d_in[14];

  char* ws = (char*)d_ws;
  __bf16*   cw    = (__bf16*)(ws + OFF_CW);
  float*    cb    = (float*)(ws + OFF_CBIAS);
  __bf16*   wwB   = (__bf16*)(ws + OFF_WWIH);
  float*    wb    = (float*)(ws + OFF_WBIAS);
  __bf16*   clsB  = (__bf16*)(ws + OFF_CLSW);
  __bf16*   feats = (__bf16*)(ws + OFF_FEATS);
  float*    wpre  = (float*)(ws + OFF_WPRE);
  float*    hs    = (float*)(ws + OFF_HS);
  __bf16*   hsb   = (__bf16*)(ws + OFF_HSB);
  unsigned* ctr   = (unsigned*)(ws + OFF_CTR);

  (void)in_sizes; (void)n_in; (void)out_size; (void)ws_size;

  (void)hipMemsetAsync(ctr, 0, 256, stream);

  k_prep_charw<<<(512 * 192 + 255) / 256, 256, 0, stream>>>(cWih, cWhh, cbih, cbhh, cw, cb);
  k_prep_wordw<<<(2048 * 384 + 255) / 256, 256, 0, stream>>>(wWih, wbih, wbhh, wwB, wb);
  k_prep_cls<<<(64 * 512 + 255) / 256, 256, 0, stream>>>(clsW, clsB);
  k_embed_words<<<8192, 256, 0, stream>>>(x, wemb, feats);

  // Char LSTM: 512 blocks x 16 words, ~216 KB dynamic LDS
  k_char_lstm<<<512, 256, 221248, stream>>>(chars, lens, cemb, cw, cb, feats);

  // Word-LSTM input projection: wpre[8192,2048] = feats @ wWih^T + (b_ih+b_hh)
  k_gemm_bias<<<dim3(8192 / 128, 2048 / 64), 256, 24576, stream>>>(
      feats, wwB, wb, wpre, 8192, 2048, 384, 384, 384, 2048);

  // Word LSTM recurrence: 16 persistent blocks, ~263 KB dynamic LDS
  k_word_lstm<<<16, 256, 128 * 520 * 4 + 512 * 4 + 128 * 4 + 32 * 4, stream>>>(
      wWhh, wpre, hs, hsb, ctr);

  // Classifier: d_out[8192,64] = hs @ cls_W^T + cls_b
  k_gemm_bias<<<dim3(8192 / 128, 1), 256, 24576, stream>>>(
      hsb, clsB, clsb, (float*)d_out, 8192, 64, 512, 512, 512, 64);
}
